// MultiScaleAttentionPoolFirst_86577950753286
// MI455X (gfx1250) — compile-verified, matrix operand order checked by reading, not run on hardware
//
#include <hip/hip_runtime.h>

typedef _Float16 h16;
typedef __attribute__((ext_vector_type(16))) _Float16 v16h;
typedef __attribute__((ext_vector_type(8)))  _Float16 v8h;
typedef __attribute__((ext_vector_type(8)))  float    v8f;
typedef __attribute__((ext_vector_type(4)))  int      v4i;

#ifndef __has_builtin
#define __has_builtin(x) 0
#endif
#if __has_builtin(__builtin_amdgcn_global_load_async_to_lds_b128) && \
    __has_builtin(__builtin_amdgcn_s_wait_asynccnt)
#define ASYNC_LDS 1
#else
#define ASYNC_LDS 0
#endif

#define AS1 __attribute__((address_space(1)))
#define AS3 __attribute__((address_space(3)))

#define B_   16
#define NH_  12
#define D_   768
#define C_   64
#define H_   56
#define P_   28
#define NP_  785       // real tokens
#define NPP_ 896       // padded tokens (multiple of 128)
#define NIN_ 3137
#define MP_  (B_ * NPP_)   // 14336 = 128 * 112
#define LDSK 40        // halves per LDS tile row (80B, 16B aligned, conflict-free)

// ---------------------------------------------------------------------------
// Pool (depthwise 3x3 stride2 pad1) + LayerNorm over 64 channels.
// Output f16 in (B, NPP, D); padded tokens zeroed each call.
// ---------------------------------------------------------------------------
__global__ __launch_bounds__(64)
void pool_ln_kernel(const float* __restrict__ x, const float* __restrict__ cw,
                    const float* __restrict__ g, const float* __restrict__ bb,
                    h16* __restrict__ out) {
    int bid   = blockIdx.x;
    int head  = bid % NH_;
    int token = (bid / NH_) % NPP_;
    int b     = bid / (NH_ * NPP_);
    int c     = threadIdx.x;
    size_t oidx = (size_t)(b * NPP_ + token) * D_ + head * C_ + c;

    if (token >= NP_) {              // uniform per block: safe early-out
        out[oidx] = (h16)0.f;
        return;
    }

    size_t xbase = (size_t)b * NIN_ * D_ + head * C_ + c;
    float val;
    if (token == 0) {
        val = x[xbase];
    } else {
        int t1 = token - 1;
        int oy = t1 / P_, ox = t1 % P_;
        float s = 0.f;
#pragma unroll
        for (int i = 0; i < 3; ++i) {
            int iy = 2 * oy - 1 + i;
            if (iy < 0 || iy >= H_) continue;
#pragma unroll
            for (int j = 0; j < 3; ++j) {
                int ix = 2 * ox - 1 + j;
                if (ix < 0 || ix >= H_) continue;
                s += cw[c * 9 + i * 3 + j] * x[xbase + (size_t)(1 + iy * H_ + ix) * D_];
            }
        }
        val = s;
    }

    __shared__ float red[64];
    red[c] = val;
    __syncthreads();
    for (int o = 32; o > 0; o >>= 1) {
        if (c < o) red[c] += red[c + o];
        __syncthreads();
    }
    float mean = red[0] * (1.f / 64.f);
    __syncthreads();
    float d = val - mean;
    red[c] = d * d;
    __syncthreads();
    for (int o = 32; o > 0; o >>= 1) {
        if (c < o) red[c] += red[c + o];
        __syncthreads();
    }
    float var = red[0] * (1.f / 64.f);
    out[oidx] = (h16)(d * rsqrtf(var + 1e-5f) * g[c] + bb[c]);
}

// ---------------------------------------------------------------------------
// Weight convert + transpose: Wt16[n*768 + k] = (f16)W[k*768 + n].
// ---------------------------------------------------------------------------
__global__ void cvt_wT_kernel(const float* __restrict__ in, h16* __restrict__ out) {
    int i = blockIdx.x * blockDim.x + threadIdx.x;
    if (i < D_ * D_) {
        int n = i / D_, k = i % D_;
        out[i] = (h16)in[k * D_ + n];
    }
}

// ---------------------------------------------------------------------------
// WMMA GEMM, tile-exact dims. B is K-minor: element (k,n) at Bt[n*ldb + k].
// Block 256 threads = 8 waves (4 M x 2 N), wave tile 32x32 (4 WMMAs/k-step),
// block tile 128x64. Double-buffered LDS, software-pipelined global->LDS
// (gfx1250 async-to-LDS when available), 1 barrier/k-step.
// Batch-offset modes: 0 -> z*stride ; 1 -> (z/NH)*stride + (z%NH)*64.
// cmode 2: compact-store rows (b,tok), tok<785, into unpadded C32.
// C16: optional scaled f16 copy. C16T: optional f16 (b,h,c,tok) transpose.
// ---------------------------------------------------------------------------
__device__ __forceinline__ size_t batch_off(int mode, long long stride, int z) {
    return (mode == 0) ? (size_t)stride * z
                       : (size_t)(z / NH_) * stride + (size_t)(z % NH_) * C_;
}

__global__ __launch_bounds__(256)
void gemm_wmma_kernel(const h16* __restrict__ A, int lda, long long strideA, int amode,
                      const h16* __restrict__ Bt, int ldb, long long strideB, int bmode,
                      const float* __restrict__ bias,
                      float* __restrict__ C32, h16* __restrict__ C16, h16* __restrict__ C16T,
                      int ldc, long long strideC, int cmode,
                      int Kdim, float scale16) {
    __shared__ h16 As[2][128 * LDSK];
    __shared__ h16 Bs[2][64 * LDSK];

    int tid  = threadIdx.x;
    int lane = tid & 31;
    int wid  = tid >> 5;
    int mw   = (wid >> 1) * 32;     // wave row base in 128-row tile
    int nw   = (wid & 1) * 32;      // wave col base in 64-col tile
    int mTile = blockIdx.y * 128;
    int nTile = blockIdx.x * 64;
    int z     = blockIdx.z;

    size_t aOff = batch_off(amode, strideA, z);
    size_t bOff = batch_off(bmode, strideB, z);

    // Cooperative loader: 2 A-chunks + 1 B-chunk (16B each) per thread.
    int lrow = tid >> 2;            // 0..63
    int lseg = (tid & 3) * 8;       // halves 0,8,16,24
    const h16* aSrc0 = A  + aOff + (size_t)(mTile + lrow) * lda + lseg;
    const h16* aSrc1 = A  + aOff + (size_t)(mTile + 64 + lrow) * lda + lseg;
    const h16* bSrc  = Bt + bOff + (size_t)(nTile + lrow) * ldb + lseg;
    int aD0 = lrow * LDSK + lseg;
    int aD1 = aD0 + 64 * LDSK;
    int bD  = aD0;

    int l15 = lane & 15;
    int hs  = (lane >= 16) ? 8 : 0;

    v8f acc00 = {}, acc01 = {}, acc10 = {}, acc11 = {};

    int kt = Kdim >> 5;             // k-tiles of 32

#if ASYNC_LDS
    // Prologue: issue tile 0 into buffer 0.
    __builtin_amdgcn_global_load_async_to_lds_b128(
        (AS1 v4i*)aSrc0, (AS3 v4i*)&As[0][aD0], 0, 0);
    __builtin_amdgcn_global_load_async_to_lds_b128(
        (AS1 v4i*)aSrc1, (AS3 v4i*)&As[0][aD1], 0, 0);
    __builtin_amdgcn_global_load_async_to_lds_b128(
        (AS1 v4i*)bSrc,  (AS3 v4i*)&Bs[0][bD],  0, 0);
#else
    uint4 ra0 = *(const uint4*)aSrc0;
    uint4 ra1 = *(const uint4*)aSrc1;
    uint4 rb  = *(const uint4*)bSrc;
#endif

    for (int i = 0; i < kt; ++i) {
        int buf = i & 1;
#if ASYNC_LDS
        __builtin_amdgcn_s_wait_asynccnt(0);   // my tile-i writes done
        __syncthreads();                       // all waves: tile i in LDS; all done with i-1
        if (i + 1 < kt) {
            int kn = (i + 1) << 5;
            int nb = buf ^ 1;
            __builtin_amdgcn_global_load_async_to_lds_b128(
                (AS1 v4i*)(aSrc0 + kn), (AS3 v4i*)&As[nb][aD0], 0, 0);
            __builtin_amdgcn_global_load_async_to_lds_b128(
                (AS1 v4i*)(aSrc1 + kn), (AS3 v4i*)&As[nb][aD1], 0, 0);
            __builtin_amdgcn_global_load_async_to_lds_b128(
                (AS1 v4i*)(bSrc + kn),  (AS3 v4i*)&Bs[nb][bD],  0, 0);
        }
#else
        *(uint4*)&As[buf][aD0] = ra0;
        *(uint4*)&As[buf][aD1] = ra1;
        *(uint4*)&Bs[buf][bD]  = rb;
        if (i + 1 < kt) {
            int kn = (i + 1) << 5;
            ra0 = *(const uint4*)(aSrc0 + kn);
            ra1 = *(const uint4*)(aSrc1 + kn);
            rb  = *(const uint4*)(bSrc + kn);
        }
        __syncthreads();
#endif
        const h16* as = As[buf];
        const h16* bs = Bs[buf];
        const v8h* a0p = (const v8h*)&as[(mw + l15) * LDSK + hs];
        const v8h* a1p = (const v8h*)&as[(mw + 16 + l15) * LDSK + hs];
        const v8h* b0p = (const v8h*)&bs[(nw + l15) * LDSK + hs];
        const v8h* b1p = (const v8h*)&bs[(nw + 16 + l15) * LDSK + hs];
        v8h a00 = a0p[0], a01 = a0p[2];
        v8h a10 = a1p[0], a11 = a1p[2];
        v8h b00 = b0p[0], b01 = b0p[2];
        v8h b10 = b1p[0], b11 = b1p[2];
        v16h af0 = __builtin_shufflevector(a00, a01, 0,1,2,3,4,5,6,7,8,9,10,11,12,13,14,15);
        v16h af1 = __builtin_shufflevector(a10, a11, 0,1,2,3,4,5,6,7,8,9,10,11,12,13,14,15);
        v16h bf0 = __builtin_shufflevector(b00, b01, 0,1,2,3,4,5,6,7,8,9,10,11,12,13,14,15);
        v16h bf1 = __builtin_shufflevector(b10, b11, 0,1,2,3,4,5,6,7,8,9,10,11,12,13,14,15);

        acc00 = __builtin_amdgcn_wmma_f32_16x16x32_f16(false, af0, false, bf0, (short)0, acc00, false, false);
        acc01 = __builtin_amdgcn_wmma_f32_16x16x32_f16(false, af0, false, bf1, (short)0, acc01, false, false);
        acc10 = __builtin_amdgcn_wmma_f32_16x16x32_f16(false, af1, false, bf0, (short)0, acc10, false, false);
        acc11 = __builtin_amdgcn_wmma_f32_16x16x32_f16(false, af1, false, bf1, (short)0, acc11, false, false);
#if !ASYNC_LDS
        __syncthreads();
#endif
    }
#if ASYNC_LDS
    __syncthreads();
#endif

    // Store.
    size_t cOff = (cmode == 1) ? batch_off(1, strideC, z) : (size_t)strideC * z;
    int col0 = nTile + nw + l15;
    int col1 = col0 + 16;
    int rb0  = mTile + mw + ((lane >= 16) ? 8 : 0);
    int rb1  = rb0 + 16;

    auto emit = [&](int row, int col, float v) {
        if (bias) v += bias[col];
        if (cmode == 2) {
            int b = row / NPP_, tok = row % NPP_;
            if (tok < NP_) C32[(size_t)(b * NP_ + tok) * ldc + col] = v;
        } else {
            size_t base = cOff + (size_t)row * ldc;
            if (C32) C32[base + col] = v;
            if (C16) C16[base + col] = (h16)(v * scale16);
            if (C16T) {
                int b = row / NPP_, tok = row % NPP_;
                C16T[(((size_t)(b * NH_ + (col >> 6))) * C_ + (col & 63)) * NPP_ + tok] = (h16)v;
            }
        }
    };
#pragma unroll
    for (int r = 0; r < 8; ++r) {
        emit(rb0 + r, col0, acc00[r]);
        emit(rb0 + r, col1, acc01[r]);
        emit(rb1 + r, col0, acc10[r]);
        emit(rb1 + r, col1, acc11[r]);
    }
}

// ---------------------------------------------------------------------------
// rel_h / rel_w: out[b,h,qy,qx,k] = sum_c q[b,h,tok,c] * rel_pos[(qc-k+27)*64+c].
// ---------------------------------------------------------------------------
__global__ void rel_kernel(const float* __restrict__ Q32, const float* __restrict__ rp,
                           float* __restrict__ out, int isW) {
    int idx = blockIdx.x * blockDim.x + threadIdx.x;
    const int total = B_ * NH_ * (P_ * P_) * P_;
    if (idx >= total) return;
    int k  = idx % P_;
    int sp = (idx / P_) % (P_ * P_);
    int bh = idx / (P_ * P_ * P_);
    int h  = bh % NH_, b = bh / NH_;
    int qc = isW ? (sp % P_) : (sp / P_);
    const float* q = Q32 + (size_t)(b * NPP_ + 1 + sp) * D_ + h * C_;
    const float* r = rp + (size_t)(qc - k + P_ - 1) * C_;
    float s = 0.f;
#pragma unroll
    for (int c = 0; c < C_; ++c) s += q[c] * r[c];
    out[idx] = s;
}

// ---------------------------------------------------------------------------
// Softmax over 785 real cols (of 896) with rel-pos bias; pads -> 0.
// ---------------------------------------------------------------------------
__global__ __launch_bounds__(256)
void softmax_kernel(const float* __restrict__ S, const float* __restrict__ RH,
                    const float* __restrict__ RW, h16* __restrict__ Pr) {
    int row = blockIdx.x;
    int qi  = row % NPP_;
    int bh  = row / NPP_;
    h16* p = Pr + (size_t)bh * NPP_ * NPP_ + (size_t)qi * NPP_;
    int tid = threadIdx.x;

    if (qi >= NP_) {
        for (int col = tid; col < NPP_; col += 256) p[col] = (h16)0.f;
        return;
    }
    const float* s = S + (size_t)bh * NPP_ * NPP_ + (size_t)qi * NPP_;
    int q1 = (qi > 0) ? (qi - 1) : 0;
    const float* rh = RH + ((size_t)bh * (P_ * P_) + q1) * P_;
    const float* rw = RW + ((size_t)bh * (P_ * P_) + q1) * P_;

    __shared__ float red[256];
    float vals[4];
    float mx = -1e30f;
#pragma unroll
    for (int it = 0; it < 4; ++it) {
        int col = tid + it * 256;
        float v = -1e30f;
        if (col < NP_) {
            v = s[col];
            if (qi >= 1 && col >= 1) {
                int k1 = col - 1;
                v += rh[k1 / P_] + rw[k1 % P_];
            }
        }
        vals[it] = v;
        mx = fmaxf(mx, v);
    }
    red[tid] = mx;
    __syncthreads();
    for (int o = 128; o > 0; o >>= 1) {
        if (tid < o) red[tid] = fmaxf(red[tid], red[tid + o]);
        __syncthreads();
    }
    mx = red[0];
    __syncthreads();

    float sum = 0.f;
#pragma unroll
    for (int it = 0; it < 4; ++it) {
        int col = tid + it * 256;
        if (col < NP_) {
            float e = __expf(vals[it] - mx);
            vals[it] = e;
            sum += e;
        } else {
            vals[it] = 0.f;
        }
    }
    red[tid] = sum;
    __syncthreads();
    for (int o = 128; o > 0; o >>= 1) {
        if (tid < o) red[tid] += red[tid + o];
        __syncthreads();
    }
    float inv = 1.f / red[0];
#pragma unroll
    for (int it = 0; it < 4; ++it) {
        int col = tid + it * 256;
        if (col < NPP_) p[col] = (h16)(vals[it] * inv);
    }
}

// ---------------------------------------------------------------------------
// Residual: O16 = f16(O32 + Q32).
// ---------------------------------------------------------------------------
__global__ void residual_kernel(const float* __restrict__ O32, const float* __restrict__ Q32,
                                h16* __restrict__ O16, int n) {
    int i = blockIdx.x * blockDim.x + threadIdx.x;
    if (i < n) O16[i] = (h16)(O32[i] + Q32[i]);
}

// ---------------------------------------------------------------------------
extern "C" void kernel_launch(void* const* d_in, const int* in_sizes, int n_in,
                              void* d_out, int out_size, void* d_ws, size_t ws_size,
                              hipStream_t stream) {
    const float* x        = (const float*)d_in[0];
    const float* pool_q_w = (const float*)d_in[1];
    const float* pool_k_w = (const float*)d_in[2];
    const float* pool_v_w = (const float*)d_in[3];
    const float* nq_g = (const float*)d_in[4];
    const float* nq_b = (const float*)d_in[5];
    const float* nk_g = (const float*)d_in[6];
    const float* nk_b = (const float*)d_in[7];
    const float* nv_g = (const float*)d_in[8];
    const float* nv_b = (const float*)d_in[9];
    const float* wq = (const float*)d_in[10];
    const float* bq = (const float*)d_in[11];
    const float* wk = (const float*)d_in[12];
    const float* bk = (const float*)d_in[13];
    const float* wv = (const float*)d_in[14];
    const float* bv = (const float*)d_in[15];
    const float* proj_w = (const float*)d_in[16];
    const float* proj_b = (const float*)d_in[17];
    const float* rel_pos_h = (const float*)d_in[18];
    const float* rel_pos_w = (const float*)d_in[19];
    (void)in_sizes; (void)n_in; (void)out_size; (void)ws_size;

    char* ws = (char*)d_ws;
    size_t off = 0;
    auto carve = [&](size_t bytes) -> char* {
        char* p = ws + off;
        off = (off + bytes + 255) & ~(size_t)255;
        return p;
    };
    const size_t MD16 = (size_t)MP_ * D_ * sizeof(h16);
    const size_t MD32 = (size_t)MP_ * D_ * sizeof(float);
    const size_t W16  = (size_t)D_ * D_ * sizeof(h16);
    const size_t S32B = (size_t)B_ * NH_ * NPP_ * NPP_ * sizeof(float);
    const size_t P16B = (size_t)B_ * NH_ * NPP_ * NPP_ * sizeof(h16);
    const size_t VT16 = (size_t)B_ * NH_ * C_ * NPP_ * sizeof(h16);
    const size_t RELB = (size_t)B_ * NH_ * (P_ * P_) * P_ * sizeof(float);

    h16*   Qp16 = (h16*)carve(MD16);
    h16*   Kp16 = (h16*)carve(MD16);
    h16*   Vp16 = (h16*)carve(MD16);
    h16*   WqT  = (h16*)carve(W16);
    h16*   WkT  = (h16*)carve(W16);
    h16*   WvT  = (h16*)carve(W16);
    h16*   WpT  = (h16*)carve(W16);
    float* Q32  = (float*)carve(MD32);
    h16*   Q16  = (h16*)carve(MD16);
    h16*   K16  = (h16*)carve(MD16);
    h16*   Vt16 = (h16*)carve(VT16);
    float* S32  = (float*)carve(S32B);
    h16*   P16  = (h16*)carve(P16B);
    float* RH   = (float*)carve(RELB);
    float* RW   = (float*)carve(RELB);
    float* O32  = (float*)carve(MD32);
    h16*   O16  = (h16*)carve(MD16);

    // 1) Weight convert + transpose to K-minor f16.
    {
        int n = D_ * D_;
        int g = (n + 255) / 256;
        cvt_wT_kernel<<<g, 256, 0, stream>>>(wq, WqT);
        cvt_wT_kernel<<<g, 256, 0, stream>>>(wk, WkT);
        cvt_wT_kernel<<<g, 256, 0, stream>>>(wv, WvT);
        cvt_wT_kernel<<<g, 256, 0, stream>>>(proj_w, WpT);
    }

    // 2) Pool + LN (f16, token-padded).
    {
        int blocks = B_ * NPP_ * NH_;
        pool_ln_kernel<<<blocks, 64, 0, stream>>>(x, pool_q_w, nq_g, nq_b, Qp16);
        pool_ln_kernel<<<blocks, 64, 0, stream>>>(x, pool_k_w, nk_g, nk_b, Kp16);
        pool_ln_kernel<<<blocks, 64, 0, stream>>>(x, pool_v_w, nv_g, nv_b, Vp16);
    }

    // 3) q/k/v projections.
    {
        dim3 grid(D_ / 64, MP_ / 128, 1);
        gemm_wmma_kernel<<<grid, 256, 0, stream>>>(
            Qp16, D_, 0LL, 0, WqT, D_, 0LL, 0, bq,
            Q32, Q16, (h16*)nullptr, D_, 0LL, 0, D_, 0.125f);
        gemm_wmma_kernel<<<grid, 256, 0, stream>>>(
            Kp16, D_, 0LL, 0, WkT, D_, 0LL, 0, bk,
            (float*)nullptr, K16, (h16*)nullptr, D_, 0LL, 0, D_, 1.0f);
        gemm_wmma_kernel<<<grid, 256, 0, stream>>>(
            Vp16, D_, 0LL, 0, WvT, D_, 0LL, 0, bv,
            (float*)nullptr, (h16*)nullptr, Vt16, D_, 0LL, 0, D_, 1.0f);
    }

    // 4) Attention scores: S = (q*scale) @ k^T, batched over 192 (b,h).
    {
        dim3 grid(NPP_ / 64, NPP_ / 128, B_ * NH_);
        gemm_wmma_kernel<<<grid, 256, 0, stream>>>(
            Q16, D_, (long long)NPP_ * D_, 1,
            K16, D_, (long long)NPP_ * D_, 1,
            (const float*)nullptr,
            S32, (h16*)nullptr, (h16*)nullptr,
            NPP_, (long long)NPP_ * NPP_, 0, C_, 1.0f);
    }

    // 5) Relative-position terms (unscaled f32 q).
    {
        int total = B_ * NH_ * (P_ * P_) * P_;
        rel_kernel<<<(total + 255) / 256, 256, 0, stream>>>(Q32, rel_pos_h, RH, 0);
        rel_kernel<<<(total + 255) / 256, 256, 0, stream>>>(Q32, rel_pos_w, RW, 1);
    }

    // 6) Bias + softmax -> f16 probs (pads zeroed).
    softmax_kernel<<<B_ * NH_ * NPP_, 256, 0, stream>>>(S32, RH, RW, P16);

    // 7) out = probs @ v (v pre-transposed), batched over (b,h).
    {
        dim3 grid(C_ / 64, NPP_ / 128, B_ * NH_);
        gemm_wmma_kernel<<<grid, 256, 0, stream>>>(
            P16, NPP_, (long long)NPP_ * NPP_, 0,
            Vt16, NPP_, (long long)C_ * NPP_, 0,
            (const float*)nullptr,
            O32, (h16*)nullptr, (h16*)nullptr,
            D_, (long long)NPP_ * D_, 1, NPP_, 1.0f);
    }

    // 8) Residual: O16 = f16(O32 + Q32).
    {
        int n = MP_ * D_;
        residual_kernel<<<(n + 255) / 256, 256, 0, stream>>>(O32, Q32, O16, n);
    }

    // 9) Final projection, compact-store into d_out (f32, unpadded).
    {
        dim3 grid(D_ / 64, MP_ / 128, 1);
        gemm_wmma_kernel<<<grid, 256, 0, stream>>>(
            O16, D_, 0LL, 0, WpT, D_, 0LL, 0, proj_b,
            (float*)d_out, (h16*)nullptr, (h16*)nullptr,
            D_, 0LL, 2, D_, 1.0f);
    }
}
